// QNetEnhance_20598663152127
// MI455X (gfx1250) — matrix operand order repaired and checked by reading, not compiled, not run on hardware
//
#include <hip/hip_runtime.h>
#include <hip/hip_bf16.h>

typedef __attribute__((ext_vector_type(16))) _Float16 v16h;
typedef __attribute__((ext_vector_type(8)))  _Float16 v8h;
typedef __attribute__((ext_vector_type(8)))  float    v8f;

// ---------------- constants ----------------
enum { LB = 2, LN = 19, LH = 8, LBN = 38, LM = 2432, LMC = 608, LND = 144 };

// =====================================================================
// Implicit-GEMM 3x3 conv (reflect pad) / ConvTranspose2d(s=2,p=1,op=1)
// One wave computes a 16(cout) x 16(pixel) f32 tile with
// v_wmma_f32_16x16x32_f16.  K is packed TAP-MAJOR: k' = r*Cin + ci,
// requiring Cin % 32 == 0.  The 9-tap loop is fully unrolled (static
// reflect offsets); each inner chunk is 32 consecutive channels of one
// tap, so the B fragment is 16 loads from a pointer walking by HH --
// no guards, no divisions, no LDS, no barriers in the hot loop.
// =====================================================================
template <int MODE>   // 0 = conv reflect-pad, 1 = convT s2 p1 op1
__global__ __launch_bounds__(32)
void conv3x3_wmma(const float* __restrict__ X, const _Float16* __restrict__ Wp,
                  const float* __restrict__ bias, float* __restrict__ Y,
                  int Cin, int Cout, int Hin, int Hout, int relu)
{
    const int lane = threadIdx.x;
    const int HWo  = Hout * Hout;
    const int HH   = Hin * Hin;
    const int p0   = blockIdx.x * 16;
    const int co0  = blockIdx.y * 16;
    const int s    = blockIdx.z;
    const int K    = Cin * 9;

    const float* Xs = X + (size_t)s * Cin * HH;
    float*       Ys = Y + (size_t)s * Cout * HWo;

    const int n  = lane & 15;           // pixel column / cout row within tile
    const int hi = lane >> 4;           // half-wave select
    const bool pv = (p0 + n) < HWo;
    const int p  = pv ? (p0 + n) : (HWo - 1);   // clamp: tail lanes load valid data
    const int oy = p / Hout, ox = p - oy * Hout;

    const _Float16* Arow = Wp + (size_t)(co0 + n) * K;
    const int a0  = hi ? 16 : 0;        // A elems 0..7  -> k = k0+a0+e
    const int a1o = hi ? 24 : 8;        // A elems 8..15 -> k = k0+a1o+(e-8)
    const int bb  = hi ? 16 : 0;        // B elems       -> ci = c0+bb+e
    const size_t stepC = (size_t)HH * 32;

    v8f acc = {};
#pragma unroll
    for (int r = 0; r < 9; ++r) {
        const int kh = r / 3, kw = r % 3;        // compile-time
        int o;
        bool vld = true;
        if (MODE == 0) {                         // reflect pad 1
            int iy = oy + kh - 1, ix = ox + kw - 1;
            iy = (iy < 0) ? -iy : ((iy >= Hin) ? 2 * Hin - 2 - iy : iy);
            ix = (ix < 0) ? -ix : ((ix >= Hin) ? 2 * Hin - 2 - ix : ix);
            o = iy * Hin + ix;
        } else {                                 // convT parity taps
            int iy2 = oy + 1 - kh, ix2 = ox + 1 - kw;
            vld = !((iy2 | ix2) & 1) && iy2 >= 0 && ix2 >= 0;
            int iy = iy2 >> 1, ix = ix2 >> 1;
            vld = vld && (iy < Hin) && (ix < Hin);
            o = vld ? (iy * Hin + ix) : 0;
        }

        const float*    qb = Xs + o + (size_t)bb * HH;
        const _Float16* Ar = Arow + r * Cin;

        for (int c0 = 0; c0 < Cin; c0 += 32) {
            // ---- A fragment: two unguarded 16B global loads ----
            v8h alo = *reinterpret_cast<const v8h*>(Ar + c0 + a0);
            v8h ahi = *reinterpret_cast<const v8h*>(Ar + c0 + a1o);
            v16h a  = __builtin_shufflevector(alo, ahi,
                        0,1,2,3,4,5,6,7,8,9,10,11,12,13,14,15);

            // ---- B fragment: 16 consecutive channels of tap r ----
            v16h b = {};
            if (MODE == 0 || vld) {
                const float* q = qb;
#pragma unroll
                for (int e = 0; e < 16; ++e) { b[e] = (_Float16)(*q); q += HH; }
            }
            qb += stepC;

            acc = __builtin_amdgcn_wmma_f32_16x16x32_f16(false, a, false, b,
                                                         (short)0, acc, false, false);
        }
    }

    // ---- writeback: VGPR r holds M=r (lanes 0-15) / M=8+r (lanes 16-31) ----
    if (pv) {
#pragma unroll
        for (int r = 0; r < 8; ++r) {
            int mrow = r + (hi ? 8 : 0);
            float v = acc[r] + bias[co0 + mrow];
            if (relu) v = fmaxf(v, 0.f);
            Ys[(size_t)(co0 + mrow) * HWo + p0 + n] = v;
        }
    }
}

// ---------------- scalar conv (a1 Cin=3, and Cout==1 heads) ----------------
__global__ void conv3x3_scalar(const float* __restrict__ X, const float* __restrict__ W,
                               const float* __restrict__ bias, float* __restrict__ Y,
                               int Cin, int Cout, int H, int relu, int S)
{
    long total = (long)S * Cout * H * H;
    long i = (long)blockIdx.x * 256 + threadIdx.x;
    if (i >= total) return;
    int ox = (int)(i % H); long t = i / H; int oy = (int)(t % H);
    t /= H; int co = (int)(t % Cout); int s = (int)(t / Cout);
    const float* xs = X + (size_t)s * Cin * H * H;
    const float* w0 = W + (size_t)co * Cin * 9;
    float acc = bias[co];
    for (int ci = 0; ci < Cin; ++ci) {
        const float* xc = xs + (size_t)ci * H * H;
        const float* wc = w0 + ci * 9;
        for (int kh = 0; kh < 3; ++kh) {
            int iy = oy + kh - 1; iy = (iy < 0) ? -iy : ((iy >= H) ? 2 * H - 2 - iy : iy);
            for (int kw = 0; kw < 3; ++kw) {
                int ix = ox + kw - 1; ix = (ix < 0) ? -ix : ((ix >= H) ? 2 * H - 2 - ix : ix);
                acc += xc[iy * H + ix] * wc[kh * 3 + kw];
            }
        }
    }
    Y[i] = relu ? fmaxf(acc, 0.f) : acc;
}

// ---------------- weight repack to half [Cout][r*Cin+ci] (tap-major) ----------------
__global__ void pack_conv_w(const float* __restrict__ W, _Float16* __restrict__ Wp,
                            int Cin, int Cout)
{
    long n = (long)Cout * Cin * 9;
    long i = (long)blockIdx.x * 256 + threadIdx.x;
    if (i >= n) return;
    int r = (int)(i % 9); long t = i / 9; int ci = (int)(t % Cin); int co = (int)(t / Cin);
    Wp[(size_t)co * Cin * 9 + (size_t)r * Cin + ci] = (_Float16)W[i];
}
// torch ConvTranspose2d weight [Cin][Cout][3][3] -> half [Cout][r*Cin+ci]
__global__ void pack_convT_w(const float* __restrict__ W, _Float16* __restrict__ Wp,
                             int Cin, int Cout)
{
    long n = (long)Cin * Cout * 9;
    long i = (long)blockIdx.x * 256 + threadIdx.x;
    if (i >= n) return;
    int r = (int)(i % 9); long t = i / 9; int co = (int)(t % Cout); int ci = (int)(t / Cout);
    Wp[(size_t)co * Cin * 9 + (size_t)r * Cin + ci] = (_Float16)W[i];
}

// ---------------- 2x2 max pool ----------------
__global__ void maxpool2x2(const float* __restrict__ X, float* __restrict__ Y,
                           int C, int H, int S)
{
    int Ho = H >> 1;
    long total = (long)S * C * Ho * Ho;
    long i = (long)blockIdx.x * 256 + threadIdx.x;
    if (i >= total) return;
    int ox = (int)(i % Ho); long t = i / Ho; int oy = (int)(t % Ho);
    t /= Ho; int c = (int)(t % C); int s = (int)(t / C);
    const float* xs = X + ((size_t)s * C + c) * H * H;
    int iy = oy * 2, ix = ox * 2;
    Y[i] = fmaxf(fmaxf(xs[iy * H + ix], xs[iy * H + ix + 1]),
                 fmaxf(xs[(iy + 1) * H + ix], xs[(iy + 1) * H + ix + 1]));
}

// ---------------- channel concat (NCHW) ----------------
__global__ void concat_ch(const float* __restrict__ A, const float* __restrict__ Bp,
                          float* __restrict__ Y, int C1, int C2, int HW, int S)
{
    long total = (long)S * (C1 + C2) * HW;
    long i = (long)blockIdx.x * 256 + threadIdx.x;
    if (i >= total) return;
    int p = (int)(i % HW); long t = i / HW;
    int c = (int)(t % (C1 + C2)); int s = (int)(t / (C1 + C2));
    Y[i] = (c < C1) ? A[((size_t)s * C1 + c) * HW + p]
                    : Bp[((size_t)s * C2 + (c - C1)) * HW + p];
}

// ---------------- anet input: cumulative one-hot scatter + coef ----------------
__global__ void anet_build(const float* __restrict__ feats, const int* __restrict__ light,
                           float* __restrict__ A0, int m0, int MC)
{
    long total = (long)MC * 64 * LND;
    long i = (long)blockIdx.x * 256 + threadIdx.x;
    if (i >= total) return;
    int d = (int)(i % LND); long t = i / LND; int c = (int)(t % 64); int lm = (int)(t / 64);
    int m = m0 + lm;
    int x = m & 7, y = (m >> 3) & 7, bn = m >> 6;
    int b = bn / LN, n = bn % LN;
    float s = 0.f;
    int base = b * LN;
    for (int q = 0; q <= n; ++q)
        if (light[base + q] == d)
            s += feats[((size_t)(base + q) * 256 + c) * 64 + y * 8 + x];
    if (c == 0) s += 1.f - powf(0.9f, (float)(n + 1));
    A0[i] = s;
}

// ---------------- vnet input: cummean | cummax, channel0 = coef ----------------
__global__ void vnet_build(const float* __restrict__ feats, float* __restrict__ V0)
{
    long total = (long)LBN * 512 * 64;
    long i = (long)blockIdx.x * 256 + threadIdx.x;
    if (i >= total) return;
    int yx = (int)(i % 64); long t = i / 64; int ch = (int)(t % 512); int bn = (int)(t / 512);
    int b = bn / LN, n = bn % LN;
    float v;
    if (ch == 0) {
        v = 1.f - powf(0.9f, (float)(n + 1));
    } else if (ch < 256) {
        float s = 0.f;
        for (int q = 0; q <= n; ++q) s += feats[((size_t)(b * LN + q) * 256 + ch) * 64 + yx];
        v = s / (float)(n + 1);
    } else {
        int c2 = ch - 256; v = -3.4e38f;
        for (int q = 0; q <= n; ++q)
            v = fmaxf(v, feats[((size_t)(b * LN + q) * 256 + c2) * 64 + yx]);
    }
    V0[i] = v;
}

// ---------------- final: xa - mean(xa) + xv ----------------
__global__ __launch_bounds__(256)
void finalize_k(const float* __restrict__ xa, const float* __restrict__ xv,
                float* __restrict__ out)
{
    int m = blockIdx.x;
    const float* xm = xa + (size_t)m * 576;
    __shared__ float red[256];
    float s = 0.f;
    for (int j = threadIdx.x; j < 576; j += 256) s += xm[j];
    red[threadIdx.x] = s;
    __syncthreads();
    for (int w = 128; w > 0; w >>= 1) {
        if ((int)threadIdx.x < w) red[threadIdx.x] += red[threadIdx.x + w];
        __syncthreads();
    }
    float add = xv[(m >> 6) * 64 + (m & 63)] - red[0] / 576.f;
    for (int j = threadIdx.x; j < 576; j += 256)
        out[(size_t)m * 576 + j] = xm[j] + add;
}

// =====================================================================
// Host orchestration
// =====================================================================
static inline unsigned g1(long n) { return (unsigned)((n + 255) / 256); }

static void run_conv(const float* X, const _Float16* Wp, const float* bias, float* Y,
                     int Cin, int Cout, int Hin, int Hout, int mode, int relu, int S,
                     hipStream_t st)
{
    dim3 g((Hout * Hout + 15) / 16, Cout / 16, S);
    if (mode == 0)
        conv3x3_wmma<0><<<g, 32, 0, st>>>(X, Wp, bias, Y, Cin, Cout, Hin, Hout, relu);
    else
        conv3x3_wmma<1><<<g, 32, 0, st>>>(X, Wp, bias, Y, Cin, Cout, Hin, Hout, relu);
}

struct LSpec { int cin, cout, isT; };
static const LSpec LS[27] = {
    {3,32,0},{32,64,0},{64,64,0},{64,128,0},{128,128,0},{128,256,0},{256,256,0},{256,256,0}, // a1..a8
    {64,64,0},{64,64,0},{64,128,0},{128,128,0},{128,256,0},{256,256,0},                      // b1..b6
    {256,256,0},{256,128,0},{128,128,0},{128,128,0},{64,64,0},{64,1,0},                      // c1..c6
    {256,128,1},{128,64,1},{128,64,1},                                                       // t1..t3
    {512,512,0},{512,512,0},{512,512,0},{512,1,0}                                            // v1..v4
};

extern "C" void kernel_launch(void* const* d_in, const int* in_sizes, int n_in,
                              void* d_out, int out_size, void* d_ws, size_t ws_size,
                              hipStream_t stream)
{
    (void)in_sizes; (void)n_in; (void)out_size; (void)ws_size;
    // pytree (sorted-key) flattening: images, light_dirs, then per layer {b, w}
    const float* images = (const float*)d_in[0];
    const int*   light  = (const int*)d_in[1];
#define PB(i) ((const float*)d_in[2 + 2 * (i)])
#define PW(i) ((const float*)d_in[2 + 2 * (i) + 1])
    float* out = (float*)d_out;
    char*  ws  = (char*)d_ws;

    size_t off = 0;
    auto alloc = [&](size_t nbytes) -> void* {
        void* p = (void*)(ws + off);
        off = (off + nbytes + 255) & ~(size_t)255;
        return p;
    };

    // ---- pack WMMA-layer weights, tap-major half [Cout][r*Cin+ci] ----
    // skipped: 0 (a1, Cin=3 -> scalar), 19 (c6), 26 (v4)
    _Float16* Wp[27];
    for (int i = 0; i < 27; ++i) {
        if (i == 0 || i == 19 || i == 26) { Wp[i] = nullptr; continue; }
        long n = (long)LS[i].cin * LS[i].cout * 9;
        Wp[i] = (_Float16*)alloc((size_t)n * sizeof(_Float16));
        if (LS[i].isT)
            pack_convT_w<<<g1(n), 256, 0, stream>>>(PW(i), Wp[i], LS[i].cin, LS[i].cout);
        else
            pack_conv_w<<<g1(n), 256, 0, stream>>>(PW(i), Wp[i], LS[i].cin, LS[i].cout);
    }

    // ---- workspace buffers (floats) ----
    const size_t BIG = (size_t)LMC * 64 * 576;              // 22.4M fl; >= stage-1 max
    float* AC0 = (float*)alloc(BIG * 4);                    // also stage-1 ping
    float* AC1 = (float*)alloc(BIG * 4);                    // also stage-1 pong
    float* S1A = AC0, *S1B = AC1;
    float* FEATS = (float*)alloc((size_t)LBN * 256 * 64 * 4);
    float* AM0 = (float*)alloc((size_t)LMC * 64 * LND * 4);
    float* AM1 = (float*)alloc((size_t)LMC * 64 * LND * 4);
    float* AM2 = (float*)alloc((size_t)LMC * 64 * LND * 4);   // x1
    float* AP0 = (float*)alloc((size_t)LMC * 64 * 36 * 4);
    float* AK0 = (float*)alloc((size_t)LMC * 128 * 36 * 4);
    float* AK1 = (float*)alloc((size_t)LMC * 128 * 36 * 4);   // x2
    float* AK2 = (float*)alloc((size_t)LMC * 128 * 36 * 4);
    float* AP1 = (float*)alloc((size_t)LMC * 128 * 9 * 4);
    float* AQ0 = (float*)alloc((size_t)LMC * 256 * 9 * 4);
    float* AQ1 = (float*)alloc((size_t)LMC * 256 * 9 * 4);
    float* AB0 = (float*)alloc((size_t)LMC * 128 * LND * 4);
    float* AB1 = (float*)alloc((size_t)LMC * 128 * LND * 4);
    float* XA  = (float*)alloc((size_t)LM * 576 * 4);
    float* VN0 = (float*)alloc((size_t)LBN * 512 * 64 * 4);
    float* VA  = (float*)alloc((size_t)LBN * 512 * 64 * 4);
    float* VB  = (float*)alloc((size_t)LBN * 512 * 64 * 4);
    float* XV  = (float*)alloc((size_t)LBN * 64 * 4);

    // ================= stage 1: image tower (38 images) =================
    conv3x3_scalar<<<g1((long)LBN*32*128*128), 256, 0, stream>>>(
        images, PW(0), PB(0), S1A, 3, 32, 128, 1, LBN);                       // a1
    maxpool2x2<<<g1((long)LBN*32*64*64), 256, 0, stream>>>(S1A, S1B, 32, 128, LBN);
    run_conv(S1B, Wp[1], PB(1), S1A, 32, 64, 64, 64, 0, 1, LBN, stream);
    run_conv(S1A, Wp[2], PB(2), S1B, 64, 64, 64, 64, 0, 1, LBN, stream);
    maxpool2x2<<<g1((long)LBN*64*32*32), 256, 0, stream>>>(S1B, S1A, 64, 64, LBN);
    run_conv(S1A, Wp[3], PB(3), S1B, 64, 128, 32, 32, 0, 1, LBN, stream);
    run_conv(S1B, Wp[4], PB(4), S1A, 128, 128, 32, 32, 0, 1, LBN, stream);
    maxpool2x2<<<g1((long)LBN*128*16*16), 256, 0, stream>>>(S1A, S1B, 128, 32, LBN);
    run_conv(S1B, Wp[5], PB(5), S1A, 128, 256, 16, 16, 0, 1, LBN, stream);
    run_conv(S1A, Wp[6], PB(6), S1B, 256, 256, 16, 16, 0, 1, LBN, stream);
    maxpool2x2<<<g1((long)LBN*256*8*8), 256, 0, stream>>>(S1B, S1A, 256, 16, LBN);
    run_conv(S1A, Wp[7], PB(7), FEATS, 256, 256, 8, 8, 0, 1, LBN, stream);

    // ================= vnet =================
    vnet_build<<<g1((long)LBN*512*64), 256, 0, stream>>>(FEATS, VN0);
    run_conv(VN0, Wp[23], PB(23), VA, 512, 512, 8, 8, 0, 1, LBN, stream);
    run_conv(VA,  Wp[24], PB(24), VB, 512, 512, 8, 8, 0, 1, LBN, stream);
    run_conv(VB,  Wp[25], PB(25), VA, 512, 512, 8, 8, 0, 1, LBN, stream);
    conv3x3_scalar<<<g1((long)LBN*64), 256, 0, stream>>>(VA, PW(26), PB(26), XV, 512, 1, 8, 0, LBN);

    // ================= anet: 4 chunks of 608 independent samples =================
    for (int c = 0; c < 4; ++c) {
        const int m0 = c * LMC;
        anet_build<<<g1((long)LMC*64*LND), 256, 0, stream>>>(FEATS, light, AM0, m0, LMC);
        run_conv(AM0, Wp[8],  PB(8),  AM1, 64, 64, 12, 12, 0, 1, LMC, stream);   // b1
        run_conv(AM1, Wp[9],  PB(9),  AM2, 64, 64, 12, 12, 0, 1, LMC, stream);   // b2 -> x1
        maxpool2x2<<<g1((long)LMC*64*36), 256, 0, stream>>>(AM2, AP0, 64, 12, LMC);
        run_conv(AP0, Wp[10], PB(10), AK0, 64, 128, 6, 6, 0, 1, LMC, stream);    // b3
        run_conv(AK0, Wp[11], PB(11), AK1, 128, 128, 6, 6, 0, 1, LMC, stream);   // b4 -> x2
        maxpool2x2<<<g1((long)LMC*128*9), 256, 0, stream>>>(AK1, AP1, 128, 6, LMC);
        run_conv(AP1, Wp[12], PB(12), AQ0, 128, 256, 3, 3, 0, 1, LMC, stream);   // b5
        run_conv(AQ0, Wp[13], PB(13), AQ1, 256, 256, 3, 3, 0, 1, LMC, stream);   // b6
        run_conv(AQ1, Wp[20], PB(20), AK2, 256, 128, 3, 6, 1, 1, LMC, stream);   // t1 (convT)
        concat_ch<<<g1((long)LMC*256*36), 256, 0, stream>>>(AK1, AK2, AM0, 128, 128, 36, LMC);
        run_conv(AM0, Wp[14], PB(14), AM1, 256, 256, 6, 6, 0, 1, LMC, stream);   // c1
        run_conv(AM1, Wp[15], PB(15), AK0, 256, 128, 6, 6, 0, 1, LMC, stream);   // c2
        run_conv(AK0, Wp[21], PB(21), AM0, 128, 64, 6, 12, 1, 1, LMC, stream);   // t2 (convT)
        concat_ch<<<g1((long)LMC*128*LND), 256, 0, stream>>>(AM2, AM0, AB0, 64, 64, LND, LMC);
        run_conv(AB0, Wp[16], PB(16), AB1, 128, 128, 12, 12, 0, 1, LMC, stream); // c3
        run_conv(AB1, Wp[17], PB(17), AB0, 128, 128, 12, 12, 0, 1, LMC, stream); // c4
        run_conv(AB0, Wp[22], PB(22), AC0, 128, 64, 12, 24, 1, 1, LMC, stream);  // t3 (convT)
        run_conv(AC0, Wp[18], PB(18), AC1, 64, 64, 24, 24, 0, 1, LMC, stream);   // c5
        conv3x3_scalar<<<g1((long)LMC*576), 256, 0, stream>>>(AC1, PW(19), PB(19),
                                                              XA + (size_t)m0 * 576, 64, 1, 24, 0, LMC);
    }

    // ================= combine =================
    finalize_k<<<LM, 256, 0, stream>>>(XA, XV, out);
#undef PB
#undef PW
}
